// feed_forward_model_45337674777137
// MI455X (gfx1250) — matrix operand order repaired and checked by reading, not compiled, not run on hardware
//
#include <hip/hip_runtime.h>
#include <math.h>

typedef __attribute__((ext_vector_type(16))) _Float16 v16h;
typedef __attribute__((ext_vector_type(8)))  _Float16 v8h;
typedef __attribute__((ext_vector_type(8)))  float    v8f;

#define NHITS   128
#define PPAIRS  8128          // C(128,2)
#define HDIM    100
#define HP      112           // H padded to 7*16
#define KP      128           // layer-2 K padded to 4*32
#define NT      7             // N tiles of 16
#define WAVES   4             // waves per block
#define ITERS   4             // row-tiles per wave (amortize weight staging)
#define ROWS_PER_BLOCK (WAVES * ITERS * 16)   // 256

// pairs with first index < i (lexicographic upper-triangle order, N=128)
__device__ __forceinline__ int cumPairs(int i) { return (i * (255 - i)) >> 1; }

__global__ __launch_bounds__(128) void pairscan_mlp_wmma(
    const float* __restrict__ x,    // [64][128][4]
    const float* __restrict__ W1,   // [8][100]
    const float* __restrict__ b1,   // [100]
    const float* __restrict__ W2,   // [100][100]
    const float* __restrict__ b2,   // [100]
    const float* __restrict__ W3,   // [100]
    const float* __restrict__ b3,   // [1]
    float* __restrict__ out)        // [64*8128]
{
    __shared__ __align__(32) _Float16 w1t[HP][32];          //  7168 B  W1^T padded
    __shared__ __align__(32) _Float16 w2t[HP][KP];          // 28672 B  W2^T padded
    __shared__ __align__(32) _Float16 htile[WAVES][16][KP]; // 16384 B per-wave h staging
    __shared__ float b1s[HP], b2s[HP], w3s[HP];

    const int tid  = threadIdx.x;
    const int wave = tid >> 5;
    const int lane = tid & 31;
    const int l16  = lane & 15;
    const bool hi  = lane >= 16;

    // ---- stage padded transposed weights into LDS (f16), once per block ----
    for (int idx = tid; idx < HP * 32; idx += 128) {
        int n = idx >> 5, k = idx & 31;
        float v = (n < HDIM && k < 8) ? W1[k * HDIM + n] : 0.0f;
        w1t[n][k] = (_Float16)v;
    }
    for (int idx = tid; idx < HP * KP; idx += 128) {
        int n = idx >> 7, k = idx & 127;
        float v = (n < HDIM && k < HDIM) ? W2[k * HDIM + n] : 0.0f;
        w2t[n][k] = (_Float16)v;
    }
    for (int n = tid; n < HP; n += 128) {
        b1s[n] = (n < HDIM) ? b1[n] : 0.0f;
        b2s[n] = (n < HDIM) ? b2[n] : 0.0f;
        w3s[n] = (n < HDIM) ? W3[n] : 0.0f;
    }
    __syncthreads();

    const float4* xv = (const float4*)x;      // x[b][n][0..3] = one float4
    const float b3v = b3[0];
    const int kloB = hi ? 16 : 0;             // B frag: lanes 0-15 K=0..15, hi K=16..31

    for (int it = 0; it < ITERS; ++it) {
        // ---- this wave's 16 rows for this iteration ----
        const int rowbase = blockIdx.x * ROWS_PER_BLOCK + it * (WAVES * 16) + wave * 16;
        const int r = rowbase + l16;          // row handled by this lane pair
        const int bidx = r / PPAIRS;
        const int p    = r - bidx * PPAIRS;

        // invert triangular index: first hit i, second hit j
        int i = (int)((255.0f - sqrtf(65025.0f - 8.0f * (float)p)) * 0.5f);
        if (i < 0) i = 0;
        while (i < NHITS - 2 && cumPairs(i + 1) <= p) ++i;
        while (i > 0 && cumPairs(i) > p) --i;
        const int j = p - cumPairs(i) + i + 1;

        float4 xa = xv[bidx * NHITS + i];
        float4 xb = xv[bidx * NHITS + j];

        // ---- A fragment for layer 1: 16x32 f16, K=0..7 real, rest zero pad ----
        // lanes 0-15 hold M=lane, elems 0..7 = K 0..7; lanes 16-31 hold K=16..31 (pad)
        v16h a1 = {};
        if (!hi) {
            a1[0] = (_Float16)xa.x; a1[1] = (_Float16)xa.y;
            a1[2] = (_Float16)xa.z; a1[3] = (_Float16)xa.w;
            a1[4] = (_Float16)xb.x; a1[5] = (_Float16)xb.y;
            a1[6] = (_Float16)xb.z; a1[7] = (_Float16)xb.w;
        }

        // ---- layer 1: h1 = pairs @ W1 + b1  (7 WMMAs) ----
        v8f acc1[NT];
#pragma unroll
        for (int t = 0; t < NT; ++t) {
            int n = t * 16 + l16;
            float bb = b1s[n];
            v8f c = {bb, bb, bb, bb, bb, bb, bb, bb};
            v16h bf = *(const v16h*)&w1t[n][kloB];
            acc1[t] = __builtin_amdgcn_wmma_f32_16x16x32_f16(
                false, a1, false, bf, (short)0, c, false, false);
        }

        // ---- stage h1 as f16 into per-wave LDS tile (cols 112..127 zeroed) ----
        if (!hi) {
            float4 z = {0.f, 0.f, 0.f, 0.f};
            *(float4*)&htile[wave][l16][112] = z;
            *(float4*)&htile[wave][l16][120] = z;
        }
#pragma unroll
        for (int t = 0; t < NT; ++t) {
            int n = t * 16 + l16;
            int mbase = hi ? 8 : 0;
#pragma unroll
            for (int v = 0; v < 8; ++v)
                htile[wave][mbase + v][n] = (_Float16)acc1[t][v];
        }
        __syncthreads();   // uniform: orders wave-local h store -> A2 load

        // ---- layer 2: h2 = h1 @ W2 + b2  (4 K-chunks x 7 N-tiles = 28 WMMAs) ----
        v8f acc2[NT];
#pragma unroll
        for (int t = 0; t < NT; ++t) {
            float bb = b2s[t * 16 + l16];
            v8f c = {bb, bb, bb, bb, bb, bb, bb, bb};
            acc2[t] = c;
        }
#pragma unroll
        for (int ck = 0; ck < 4; ++ck) {
            // A fragment from h tile: lane row M=l16; elems 0..7 = K klo..klo+7,
            // elems 8..15 = K klo+8..klo+15 (hi lanes shifted by 16)
            int klo = ck * 32 + (hi ? 16 : 0);
            v8h alo = *(const v8h*)&htile[wave][l16][klo];
            v8h ahi = *(const v8h*)&htile[wave][l16][klo + 8];
            v16h a2;
#pragma unroll
            for (int e = 0; e < 8; ++e) { a2[e] = alo[e]; a2[e + 8] = ahi[e]; }

            int kb = ck * 32 + (hi ? 16 : 0);
#pragma unroll
            for (int t = 0; t < NT; ++t) {
                int n = t * 16 + l16;
                v16h bf = *(const v16h*)&w2t[n][kb];
                acc2[t] = __builtin_amdgcn_wmma_f32_16x16x32_f16(
                    false, a2, false, bf, (short)0, acc2[t], false, false);
            }
        }

        // ---- layer 3 + sigmoid: per-row dot(relu(h2), W3) + b3 ----
#pragma unroll
        for (int v = 0; v < 8; ++v) {
            float s = 0.0f;
#pragma unroll
            for (int t = 0; t < NT; ++t)
                s += fmaxf(acc2[t][v], 0.0f) * w3s[t * 16 + l16];
            // reduce across the 16 lanes of each half-wave (wave32)
            s += __shfl_xor(s, 8, 32);
            s += __shfl_xor(s, 4, 32);
            s += __shfl_xor(s, 2, 32);
            s += __shfl_xor(s, 1, 32);
            if (l16 == 0) {
                int M = v + (hi ? 8 : 0);      // lane 0 -> row v, lane 16 -> row v+8
                float z = s + b3v;
                out[rowbase + M] = 1.0f / (1.0f + __expf(-z));
            }
        }
        __syncthreads();   // uniform: WAR fence before next iteration's h stores
    }
}

extern "C" void kernel_launch(void* const* d_in, const int* in_sizes, int n_in,
                              void* d_out, int out_size, void* d_ws, size_t ws_size,
                              hipStream_t stream) {
    const float* x  = (const float*)d_in[0];
    // d_in[1] = in_hitnr (unused by the reference computation)
    const float* W1 = (const float*)d_in[2];
    const float* b1 = (const float*)d_in[3];
    const float* W2 = (const float*)d_in[4];
    const float* b2 = (const float*)d_in[5];
    const float* W3 = (const float*)d_in[6];
    const float* b3 = (const float*)d_in[7];
    float* out = (float*)d_out;

    dim3 grid(2032);   // 64*8128 rows / 256 rows per block
    dim3 block(128);   // 4 waves of 32
    hipLaunchKernelGGL(pairscan_mlp_wmma, grid, block, 0, stream,
                       x, W1, b1, W2, b2, W3, b3, out);
}
